// MoELayer_51625506898425
// MI455X (gfx1250) — matrix operand order repaired and checked by reading, not compiled, not run on hardware
//
#include <hip/hip_runtime.h>
#include <math.h>

// Problem constants (from reference): B=4, S=2048 -> T=8192 tokens
#define TOK   8192
#define EDIM  1024
#define HDIM  2048
#define NEXP  8
#define MT    16          // token tile per workgroup
#define BLK   256         // 8 waves of 32

typedef __attribute__((ext_vector_type(16))) __bf16 v16bf;
typedef __attribute__((ext_vector_type(8)))  __bf16 v8bf;
typedef __attribute__((ext_vector_type(4)))  __bf16 v4bf;
typedef __attribute__((ext_vector_type(8)))  float  v8f;

// float -> bf16, round-to-nearest-even
__device__ __forceinline__ __bf16 f2bf(float f) {
  unsigned u = __builtin_bit_cast(unsigned, f);
  u += 0x7FFFu + ((u >> 16) & 1u);
  unsigned short s = (unsigned short)(u >> 16);
  return __builtin_bit_cast(__bf16, s);
}

__device__ __forceinline__ float gelu_exact(float v) {
  return 0.5f * v * (1.0f + erff(v * 0.70710678118654752f));
}

// A fragment: 16x32 bf16 from LDS, contiguous K layout (matches B):
//  lanes 0-15 : elems 0-15 = K kk..kk+15     (row M = lane)
//  lanes 16-31: elems 0-15 = K kk+16..kk+31  (row M = lane-16)
__device__ __forceinline__ v16bf load_a_frag(const __bf16* h, int ld, int kk, int lane) {
  int row  = lane & 15;
  int koff = kk + ((lane >> 4) << 4);
  const __bf16* p = h + row * ld + koff;       // 16B aligned (kk multiple of 32)
  v8bf lo = *reinterpret_cast<const v8bf*>(p);
  v8bf hi = *reinterpret_cast<const v8bf*>(p + 8);
  return __builtin_shufflevector(lo, hi, 0,1,2,3,4,5,6,7,8,9,10,11,12,13,14,15);
}

// B fragment: 32x16 from PRE-TRANSPOSED bf16 weights Wt[N][K] (K contiguous):
//  lanes 0-15 hold col N=n0+lane,    K=kk..kk+15
//  lanes 16-31 hold col N=n0+lane-16, K=kk+16..kk+31
// -> 16 contiguous bf16 per lane = two global_load_b128.
__device__ __forceinline__ v16bf load_b_frag_t(const __bf16* __restrict__ Wt, int ldk,
                                               int kk, int n0, int lane) {
  int col   = n0 + (lane & 15);
  int kbase = kk + ((lane >> 4) << 4);
  const __bf16* p = Wt + (size_t)col * ldk + kbase;   // 32B aligned
  __builtin_prefetch(p + 32, 0, 1);                   // next K-slab (speculative, drop-safe)
  v8bf lo = *reinterpret_cast<const v8bf*>(p);
  v8bf hi = *reinterpret_cast<const v8bf*>(p + 8);
  return __builtin_shufflevector(lo, hi, 0,1,2,3,4,5,6,7,8,9,10,11,12,13,14,15);
}

__device__ __forceinline__ v8f wmma_bf16(v16bf a, v16bf b, v8f c) {
  return __builtin_amdgcn_wmma_f32_16x16x32_bf16(false, a, false, b,
                                                 (short)0, c, false, false);
}

// ---------------- Weight convert+transpose: f32 [K][N] -> bf16 [N][K] -----------
// One 32x32 tile per block; coalesced float4 reads, coalesced 8B bf16 writes.
__launch_bounds__(BLK)
__global__ void transpose_bf16_kernel(const float* __restrict__ src,
                                      __bf16* __restrict__ dst, int K, int N) {
  __shared__ __bf16 tile[32][33];                     // padded to dodge bank conflicts
  const int e  = blockIdx.y;
  const int nt = N >> 5;
  const int tk = blockIdx.x / nt, tn = blockIdx.x % nt;
  const int r  = threadIdx.x >> 3;
  const int cq = (threadIdx.x & 7) << 2;

  const float* s = src + (size_t)e * K * N + (size_t)(tk * 32 + r) * N + tn * 32 + cq;
  float4 v = *reinterpret_cast<const float4*>(s);
  tile[r][cq + 0] = f2bf(v.x);
  tile[r][cq + 1] = f2bf(v.y);
  tile[r][cq + 2] = f2bf(v.z);
  tile[r][cq + 3] = f2bf(v.w);
  __syncthreads();

  v4bf o;
  o[0] = tile[cq + 0][r];
  o[1] = tile[cq + 1][r];
  o[2] = tile[cq + 2][r];
  o[3] = tile[cq + 3][r];
  __bf16* d = dst + (size_t)e * N * K + (size_t)(tn * 32 + r) * K + tk * 32 + cq;
  *reinterpret_cast<v4bf*>(d) = o;                    // 8B store, aligned
}

// ---------------- Gate: logits -> softmax -> top-2 -> dense selection weights ----
__launch_bounds__(BLK)
__global__ void gate_topk_kernel(const float* __restrict__ x,
                                 const float* __restrict__ gate_w,  // [E, NE]
                                 const float* __restrict__ gate_b,  // [NE]
                                 float* __restrict__ gsel) {        // [T, NE]
  const int token = blockIdx.x * (BLK / 32) + (threadIdx.x >> 5);
  const int lane  = threadIdx.x & 31;
  const float* xr = x + (size_t)token * EDIM;

  float l[NEXP];
#pragma unroll
  for (int n = 0; n < NEXP; ++n) l[n] = 0.f;

  for (int c = lane; c < EDIM; c += 32) {
    float xv = xr[c];
    const float4* g4 = reinterpret_cast<const float4*>(gate_w + (size_t)c * NEXP);
    float4 g0 = g4[0], g1 = g4[1];
    l[0] += xv * g0.x; l[1] += xv * g0.y; l[2] += xv * g0.z; l[3] += xv * g0.w;
    l[4] += xv * g1.x; l[5] += xv * g1.y; l[6] += xv * g1.z; l[7] += xv * g1.w;
  }
#pragma unroll
  for (int off = 16; off > 0; off >>= 1) {
#pragma unroll
    for (int n = 0; n < NEXP; ++n) l[n] += __shfl_xor(l[n], off, 32);
  }

  if (lane == 0) {
#pragma unroll
    for (int n = 0; n < NEXP; ++n) l[n] += gate_b[n];
    float m = l[0];
#pragma unroll
    for (int n = 1; n < NEXP; ++n) m = fmaxf(m, l[n]);
    float p[NEXP], s = 0.f;
#pragma unroll
    for (int n = 0; n < NEXP; ++n) { p[n] = expf(l[n] - m); s += p[n]; }
    float inv = 1.0f / s;
#pragma unroll
    for (int n = 0; n < NEXP; ++n) p[n] *= inv;
    // top-2, first-index tie-break (matches jax.lax.top_k)
    int   i1 = 0; float v1 = p[0];
#pragma unroll
    for (int n = 1; n < NEXP; ++n) if (p[n] > v1) { v1 = p[n]; i1 = n; }
    int   i2 = -1; float v2 = -1.f;
#pragma unroll
    for (int n = 0; n < NEXP; ++n)
      if (n != i1 && p[n] > v2) { v2 = p[n]; i2 = n; }
    float* g = gsel + (size_t)token * NEXP;
#pragma unroll
    for (int n = 0; n < NEXP; ++n)
      g[n] = (n == i1) ? v1 : ((n == i2) ? v2 : 0.f);
  }
}

// ---------------- Fused 3-GEMM expert FFN, WMMA bf16, LDS-resident h1/h2 --------
__launch_bounds__(BLK)
__global__ void moe_ffn_kernel(const float* __restrict__ x,
                               const __bf16* __restrict__ w1t,   // [NE][H][E]
                               const float*  __restrict__ b1,
                               const __bf16* __restrict__ w2t,   // [NE][H][H]
                               const float*  __restrict__ b2,
                               const __bf16* __restrict__ w3t,   // [NE][E][H]
                               const float*  __restrict__ b3,
                               const float*  __restrict__ gsel,
                               float* __restrict__ out) {
  extern __shared__ __align__(16) __bf16 smem[];
  __bf16* lds_x  = smem;                              // MT * EDIM  (32 KB)
  __bf16* lds_h1 = smem + MT * EDIM;                  // MT * HDIM  (64 KB)
  __bf16* lds_h2 = smem + MT * EDIM + MT * HDIM;      // MT * HDIM  (64 KB)

  const int t0     = blockIdx.x * MT;
  const int wave   = threadIdx.x >> 5;
  const int lane   = threadIdx.x & 31;
  const int rowoff = (lane >> 4) << 3;                // C/D: rows 0-7 or 8-15
  const int col16  = lane & 15;

  // Stage x tile, f32 -> bf16 (float4 in, packed 8B ds_store out)
  for (int idx = threadIdx.x; idx < MT * EDIM / 4; idx += BLK) {
    int r  = idx / (EDIM / 4);
    int c4 = idx % (EDIM / 4);
    const float4 v = reinterpret_cast<const float4*>(x + (size_t)(t0 + r) * EDIM)[c4];
    v4bf o; o[0] = f2bf(v.x); o[1] = f2bf(v.y); o[2] = f2bf(v.z); o[3] = f2bf(v.w);
    *reinterpret_cast<v4bf*>(&lds_x[r * EDIM + c4 * 4]) = o;
  }
  __syncthreads();

  v8f zero;
#pragma unroll
  for (int j = 0; j < 8; ++j) zero[j] = 0.f;

  v8f acc[8];                        // persistent out accum: wave owns cols [wave*128, +128)
#pragma unroll
  for (int i = 0; i < 8; ++i) acc[i] = zero;

  for (int e = 0; e < NEXP; ++e) {
    // ---- GEMM1: h1 = gelu(x @ w1[e] + b1[e])   [16x1024]x[1024x2048]
    {
      const __bf16* W1 = w1t + (size_t)e * HDIM * EDIM;
      const float*  B1 = b1  + (size_t)e * HDIM;
#pragma unroll 1
      for (int nt = wave * 16; nt < wave * 16 + 16; nt += 2) {
        int n0 = nt * 16, n1 = n0 + 16;
        v8f c0 = zero, c1 = zero;
        for (int kk = 0; kk < EDIM; kk += 32) {
          v16bf a  = load_a_frag(lds_x, EDIM, kk, lane);
          v16bf bm0 = load_b_frag_t(W1, EDIM, kk, n0, lane);
          v16bf bm1 = load_b_frag_t(W1, EDIM, kk, n1, lane);
          c0 = wmma_bf16(a, bm0, c0);
          c1 = wmma_bf16(a, bm1, c1);
        }
        float bias0 = B1[n0 + col16], bias1 = B1[n1 + col16];
#pragma unroll
        for (int v = 0; v < 8; ++v) {
          lds_h1[(v + rowoff) * HDIM + n0 + col16] = f2bf(gelu_exact(c0[v] + bias0));
          lds_h1[(v + rowoff) * HDIM + n1 + col16] = f2bf(gelu_exact(c1[v] + bias1));
        }
      }
      __syncthreads();
    }
    // ---- GEMM2: h2 = gelu(h1 @ w2[e] + b2[e])  [16x2048]x[2048x2048]
    {
      const __bf16* W2 = w2t + (size_t)e * HDIM * HDIM;
      const float*  B2 = b2  + (size_t)e * HDIM;
#pragma unroll 1
      for (int nt = wave * 16; nt < wave * 16 + 16; nt += 2) {
        int n0 = nt * 16, n1 = n0 + 16;
        v8f c0 = zero, c1 = zero;
        for (int kk = 0; kk < HDIM; kk += 32) {
          v16bf a   = load_a_frag(lds_h1, HDIM, kk, lane);
          v16bf bm0 = load_b_frag_t(W2, HDIM, kk, n0, lane);
          v16bf bm1 = load_b_frag_t(W2, HDIM, kk, n1, lane);
          c0 = wmma_bf16(a, bm0, c0);
          c1 = wmma_bf16(a, bm1, c1);
        }
        float bias0 = B2[n0 + col16], bias1 = B2[n1 + col16];
#pragma unroll
        for (int v = 0; v < 8; ++v) {
          lds_h2[(v + rowoff) * HDIM + n0 + col16] = f2bf(gelu_exact(c0[v] + bias0));
          lds_h2[(v + rowoff) * HDIM + n1 + col16] = f2bf(gelu_exact(c1[v] + bias1));
        }
      }
      __syncthreads();
    }
    // ---- GEMM3: out_e = h2 @ w3[e] + b3[e], scaled by gate weight, accumulated
    {
      const __bf16* W3 = w3t + (size_t)e * EDIM * HDIM;
      const float*  B3 = b3  + (size_t)e * EDIM;
      float gw[8];
#pragma unroll
      for (int v = 0; v < 8; ++v)
        gw[v] = gsel[(size_t)(t0 + v + rowoff) * NEXP + e];
#pragma unroll 1
      for (int nt = 0; nt < 8; nt += 2) {
        int n0 = wave * 128 + nt * 16, n1 = n0 + 16;
        v8f c0 = zero, c1 = zero;
        for (int kk = 0; kk < HDIM; kk += 32) {
          v16bf a   = load_a_frag(lds_h2, HDIM, kk, lane);
          v16bf bm0 = load_b_frag_t(W3, HDIM, kk, n0, lane);
          v16bf bm1 = load_b_frag_t(W3, HDIM, kk, n1, lane);
          c0 = wmma_bf16(a, bm0, c0);
          c1 = wmma_bf16(a, bm1, c1);
        }
        float bias0 = B3[n0 + col16], bias1 = B3[n1 + col16];
#pragma unroll
        for (int v = 0; v < 8; ++v) {
          acc[nt + 0][v] += gw[v] * (c0[v] + bias0);
          acc[nt + 1][v] += gw[v] * (c1[v] + bias1);
        }
      }
      __syncthreads();   // lds_h1/h2 reused next expert
    }
  }

  // Write final output (single owner per element -> deterministic)
#pragma unroll 1
  for (int nt = 0; nt < 8; ++nt) {
    int col = wave * 128 + nt * 16 + col16;
#pragma unroll
    for (int v = 0; v < 8; ++v)
      out[(size_t)(t0 + v + rowoff) * EDIM + col] = acc[nt][v];
  }
}

extern "C" void kernel_launch(void* const* d_in, const int* in_sizes, int n_in,
                              void* d_out, int out_size, void* d_ws, size_t ws_size,
                              hipStream_t stream) {
  (void)in_sizes; (void)n_in; (void)out_size; (void)ws_size;
  const float* x      = (const float*)d_in[0];
  const float* w1     = (const float*)d_in[1];
  const float* b1     = (const float*)d_in[2];
  const float* w2     = (const float*)d_in[3];
  const float* b2     = (const float*)d_in[4];
  const float* w3     = (const float*)d_in[5];
  const float* b3     = (const float*)d_in[6];
  const float* gate_w = (const float*)d_in[7];
  const float* gate_b = (const float*)d_in[8];
  float* out = (float*)d_out;

  // Workspace layout: gsel [T][NE] f32, then bf16 transposed weights.
  char* ws = (char*)d_ws;
  float*  gsel = (float*)ws;                                   //   256 KB
  __bf16* w1t  = (__bf16*)(ws + (size_t)TOK * NEXP * 4);       //  33.5 MB  [NE][H][E]
  __bf16* w2t  = w1t + (size_t)NEXP * HDIM * EDIM;             //  67.1 MB  [NE][H][H]
  __bf16* w3t  = w2t + (size_t)NEXP * HDIM * HDIM;             //  33.5 MB  [NE][E][H]

  // 1) gate softmax + top-2
  gate_topk_kernel<<<TOK / (BLK / 32), BLK, 0, stream>>>(x, gate_w, gate_b, gsel);

  // 2) weight convert+transpose (f32 [K][N] -> bf16 [N][K]) — one-time pass
  transpose_bf16_kernel<<<dim3((EDIM / 32) * (HDIM / 32), NEXP), BLK, 0, stream>>>(
      w1, w1t, EDIM, HDIM);
  transpose_bf16_kernel<<<dim3((HDIM / 32) * (HDIM / 32), NEXP), BLK, 0, stream>>>(
      w2, w2t, HDIM, HDIM);
  transpose_bf16_kernel<<<dim3((HDIM / 32) * (EDIM / 32), NEXP), BLK, 0, stream>>>(
      w3, w3t, HDIM, EDIM);

  // 3) fused expert FFN
  const size_t lds_bytes = (size_t)(MT * EDIM + 2 * MT * HDIM) * sizeof(__bf16); // 160 KB
  moe_ffn_kernel<<<TOK / MT, BLK, lds_bytes, stream>>>(x, w1t, b1, w2t, b2, w3t, b3,
                                                       gsel, out);
}
